// Model_9955734192568
// MI455X (gfx1250) — compile-verified, matrix operand order checked by reading, not run on hardware
//
#include <hip/hip_runtime.h>
#include <hip/hip_bf16.h>

// ---------------------------------------------------------------------------
// VQ-VAE forward for MI455X (gfx1250, wave32, WMMA).
// GEMMs run on v_wmma_f32_16x16x32_bf16 (f32 accumulate); the 512MB
// all_vq_z broadcast uses non-temporal 128-bit stores.
// ---------------------------------------------------------------------------

typedef __attribute__((ext_vector_type(16))) __bf16   v16bf;
typedef __attribute__((ext_vector_type(8)))  float    v8f;
typedef __attribute__((ext_vector_type(4)))  unsigned v4u;
typedef __attribute__((ext_vector_type(4)))  float    v4f;

#define B_SZ   4096
#define D_SZ   512
#define K_SZ   64
#define H_SZ   4
#define HID_SZ 512
#define CC_F   0.25f

__device__ __forceinline__ unsigned short f2bf(float f) {
  unsigned u = __float_as_uint(f);
  return (unsigned short)((u + 0x7FFFu + ((u >> 16) & 1u)) >> 16); // RNE
}
__device__ __forceinline__ float bf2f(unsigned short s) {
  return __uint_as_float(((unsigned)s) << 16);
}

union FragU { v4u u[2]; v16bf v; };

// A-fragment layout (ISA 7.12.2, 16-bit A 16x32): lane<16 holds row (lane),
// K = {0..7, 16..23}; lane>=16 holds row (lane-16), K = {8..15, 24..31}.
// Two 16B chunks per lane. B-matrix loaded from B^T with identical pattern.
__device__ __forceinline__ v16bf load_frag(const unsigned short* __restrict__ base,
                                           int ld, int row0, int k0, int lr, int hi) {
  const unsigned short* p = base + (size_t)(row0 + lr) * ld + (k0 + hi * 8);
  FragU f;
  f.u[0] = *reinterpret_cast<const v4u*>(p);
  f.u[1] = *reinterpret_cast<const v4u*>(p + 16);
  return f.v;
}

// ---------------------------------------------------------------------------
// Generic bf16 WMMA GEMM: C[M,N] = A[M,Kd] * Bt[N,Kd]^T, 64x64 block,
// 4 waves, each wave owns a 32x32 tile (2x2 WMMA accumulators).
// EPI 0: out_bf16 = relu(acc + coladd[n])            (GEMM1 -> h)
// EPI 1: out_bf16 = acc + coladd[n]                  (GEMM2 -> z)
// EPI 2: out_f32  = rowadd[m] + coladd[n] - 2*acc    (distances)
// EPI 3: out_f32  = 1.25*(rowadd[m]+coladd[n]-2*acc) (all_vq_loss)
// ---------------------------------------------------------------------------
template <int EPI>
__global__ __launch_bounds__(128) void gemm_bf16_wmma(
    const unsigned short* __restrict__ A, const unsigned short* __restrict__ Bt,
    const float* __restrict__ rowadd, const float* __restrict__ coladd,
    float* __restrict__ outF, unsigned short* __restrict__ outB,
    int Kd, int ldA, int ldB, int ldC,
    size_t sA, size_t sB, size_t sC, size_t sRow, size_t sCol) {
  const int h = blockIdx.z;
  A  += (size_t)h * sA;
  Bt += (size_t)h * sB;
  if (rowadd) rowadd += (size_t)h * sRow;
  if (coladd) coladd += (size_t)h * sCol;
  if (outF)   outF   += (size_t)h * sC;
  if (outB)   outB   += (size_t)h * sC;

  const int lane = threadIdx.x & 31;
  const int wave = threadIdx.x >> 5;
  const int wm = wave >> 1, wn = wave & 1;
  const int m0 = blockIdx.y * 64 + wm * 32;
  const int n0 = blockIdx.x * 64 + wn * 32;
  const int lr = lane & 15, hi = lane >> 4;

  v8f acc[2][2] = {};

  for (int k0 = 0; k0 < Kd; k0 += 32) {
    v16bf a0 = load_frag(A, ldA, m0, k0, lr, hi);
    v16bf a1 = load_frag(A, ldA, m0 + 16, k0, lr, hi);
    v16bf b0 = load_frag(Bt, ldB, n0, k0, lr, hi);
    v16bf b1 = load_frag(Bt, ldB, n0 + 16, k0, lr, hi);
    acc[0][0] = __builtin_amdgcn_wmma_f32_16x16x32_bf16(false, a0, false, b0, (short)0, acc[0][0], false, false);
    acc[0][1] = __builtin_amdgcn_wmma_f32_16x16x32_bf16(false, a0, false, b1, (short)0, acc[0][1], false, false);
    acc[1][0] = __builtin_amdgcn_wmma_f32_16x16x32_bf16(false, a1, false, b0, (short)0, acc[1][0], false, false);
    acc[1][1] = __builtin_amdgcn_wmma_f32_16x16x32_bf16(false, a1, false, b1, (short)0, acc[1][1], false, false);
  }

  // C/D layout: VGPR i, lane<16 -> (M=i, N=lane); lane>=16 -> (M=i+8, N=lane-16)
#pragma unroll
  for (int mi = 0; mi < 2; ++mi) {
#pragma unroll
    for (int ni = 0; ni < 2; ++ni) {
      const int col = n0 + ni * 16 + lr;
      const float ca = coladd ? coladd[col] : 0.0f;
#pragma unroll
      for (int i = 0; i < 8; ++i) {
        const int row = m0 + mi * 16 + i + hi * 8;
        float v = acc[mi][ni][i];
        if (EPI == 0) {
          v += ca; v = v > 0.0f ? v : 0.0f;
          outB[(size_t)row * ldC + col] = f2bf(v);
        } else if (EPI == 1) {
          v += ca;
          outB[(size_t)row * ldC + col] = f2bf(v);
        } else if (EPI == 2) {
          outF[(size_t)row * ldC + col] = rowadd[row] + ca - 2.0f * v;
        } else {
          outF[(size_t)row * ldC + col] = (1.0f + CC_F) * (rowadd[row] + ca - 2.0f * v);
        }
      }
    }
  }
}

// ---------------------------------------------------------------------------
// Helpers
// ---------------------------------------------------------------------------
__global__ void cvt_f32_bf16(const float* __restrict__ in, unsigned short* __restrict__ out, size_t n) {
  size_t i = blockIdx.x * (size_t)blockDim.x + threadIdx.x;
  size_t stride = (size_t)gridDim.x * blockDim.x;
  for (; i < n; i += stride) out[i] = f2bf(in[i]);
}

// in: [H, Kd, N] row-major -> out: [H, N, Kd] bf16 (B^T layout for WMMA loads)
__global__ void transpose_cvt(const float* __restrict__ in, unsigned short* __restrict__ out,
                              int Kd, int N, size_t total) {
  size_t stride = (size_t)gridDim.x * blockDim.x;
  size_t plane = (size_t)Kd * N;
  for (size_t o = blockIdx.x * (size_t)blockDim.x + threadIdx.x; o < total; o += stride) {
    size_t h = o / plane;
    size_t r = o - h * plane;
    size_t nn = r / (size_t)Kd;
    size_t kk = r - nn * (size_t)Kd;
    out[o] = f2bf(in[h * plane + kk * (size_t)N + nn]);
  }
}

// avz[k,d] = mean_h E[h,k,d]  (f32 + bf16 copies)
__global__ void avz_kernel(const float* __restrict__ E, float* __restrict__ avz,
                           unsigned short* __restrict__ avzb, int KD) {
  int i = blockIdx.x * blockDim.x + threadIdx.x;
  if (i >= KD) return;
  float s = 0.0f;
#pragma unroll
  for (int h = 0; h < H_SZ; ++h) s += E[(size_t)h * KD + i];
  s *= (1.0f / H_SZ);
  avz[i] = s;
  avzb[i] = f2bf(s);
}

// ||row||^2 for bf16 rows of length rowlen; one wave32 per row.
__global__ __launch_bounds__(256) void rownorm_bf16(const unsigned short* __restrict__ A,
                                                    float* __restrict__ out, int rowlen) {
  const int lane = threadIdx.x & 31;
  const int row = blockIdx.x * 8 + (threadIdx.x >> 5);
  const unsigned short* p = A + (size_t)row * rowlen;
  float s = 0.0f;
  for (int i = lane; i < rowlen; i += 32) { float v = bf2f(p[i]); s += v * v; }
#pragma unroll
  for (int off = 16; off; off >>= 1) s += __shfl_xor(s, off, 32);
  if (lane == 0) out[row] = s;
}

// argmin over K=64 of head-summed distances (mean is monotonic). Lexicographic
// min (val, idx) reduction => first-min tie-break like jnp.argmin.
__global__ __launch_bounds__(256) void argmin_kernel(const float* __restrict__ dist,
                                                     int* __restrict__ idx) {
  const int lane = threadIdx.x & 31;
  const int b = blockIdx.x * 8 + (threadIdx.x >> 5);
  float v0 = 0.0f, v1 = 0.0f;
#pragma unroll
  for (int h = 0; h < H_SZ; ++h) {
    const float* p = dist + ((size_t)h * B_SZ + b) * K_SZ;
    v0 += p[lane];
    v1 += p[lane + 32];
  }
  float bv = v0; int bi = lane;
  if (v1 < bv) { bv = v1; bi = lane + 32; }
#pragma unroll
  for (int off = 16; off; off >>= 1) {
    float ov = __shfl_xor(bv, off, 32);
    int oi = __shfl_xor(bi, off, 32);
    if (ov < bv || (ov == bv && oi < bi)) { bv = ov; bi = oi; }
  }
  if (lane == 0) idx[b] = bi;
}

// vq_z[b,d] = mean_h E[h, idx[b], d]  (fp32 E for full precision)
__global__ void vq_gather(const float* __restrict__ E, const int* __restrict__ idx,
                          float* __restrict__ out) {
  size_t i = blockIdx.x * (size_t)blockDim.x + threadIdx.x;
  int b = (int)(i >> 9);          // D = 512
  int d = (int)(i & 511);
  int id = idx[b];
  float s = 0.0f;
#pragma unroll
  for (int h = 0; h < H_SZ; ++h) s += E[((size_t)h * K_SZ + id) * D_SZ + d];
  out[i] = s * (1.0f / H_SZ);
}

// all_vq_z[k,b,:] = avz[k,:]  — 512 MB broadcast, non-temporal b128 stores.
__global__ void broadcast_avz(const v4f* __restrict__ avz4, v4f* __restrict__ out4,
                              unsigned rowlen4, unsigned perK) {
  unsigned t = blockIdx.x * blockDim.x + threadIdx.x;
  unsigned k = t / perK;
  unsigned j = t % rowlen4;
  v4f v = avz4[k * rowlen4 + j];
  __builtin_nontemporal_store(v, out4 + t);
}

// ---------------------------------------------------------------------------
extern "C" void kernel_launch(void* const* d_in, const int* in_sizes, int n_in,
                              void* d_out, int out_size, void* d_ws, size_t ws_size,
                              hipStream_t stream) {
  const float* x  = (const float*)d_in[0];
  const float* W1 = (const float*)d_in[1];
  const float* b1 = (const float*)d_in[2];
  const float* W2 = (const float*)d_in[3];
  const float* b2 = (const float*)d_in[4];
  const float* E  = (const float*)d_in[5];

  // Output sections (flat, in tuple return order)
  float* out0 = (float*)d_out;                         // vq_z      [B,D]
  float* dist = out0 + (size_t)B_SZ * D_SZ;            // distances [H,B,K]
  float* avqz = dist + (size_t)H_SZ * B_SZ * K_SZ;     // all_vq_z  [K,B,D]
  float* loss = avqz + (size_t)K_SZ * B_SZ * D_SZ;     // all_vq_loss [K,B]

  // Workspace carve-up (~42 MB)
  char* ws = (char*)d_ws;
  size_t off = 0;
  auto alloc = [&](size_t bytes) { char* p = ws + off; off = (off + bytes + 255) & ~(size_t)255; return p; };
  unsigned short* xb    = (unsigned short*)alloc((size_t)B_SZ * D_SZ * 2);
  unsigned short* w1t   = (unsigned short*)alloc((size_t)H_SZ * HID_SZ * D_SZ * 2);
  unsigned short* w2t   = (unsigned short*)alloc((size_t)H_SZ * D_SZ * HID_SZ * 2);
  unsigned short* eb    = (unsigned short*)alloc((size_t)H_SZ * K_SZ * D_SZ * 2);
  float*          avzf  = (float*)alloc((size_t)K_SZ * D_SZ * 4);
  unsigned short* avzb  = (unsigned short*)alloc((size_t)K_SZ * D_SZ * 2);
  unsigned short* hb    = (unsigned short*)alloc((size_t)H_SZ * B_SZ * HID_SZ * 2);
  unsigned short* zb    = (unsigned short*)alloc((size_t)H_SZ * B_SZ * D_SZ * 2);
  float*          xnrm  = (float*)alloc((size_t)B_SZ * 4);
  float*          enrm  = (float*)alloc((size_t)H_SZ * K_SZ * 4);
  float*          aznrm = (float*)alloc((size_t)K_SZ * 4);
  float*          znrm  = (float*)alloc((size_t)H_SZ * B_SZ * 4);
  int*            idxp  = (int*)alloc((size_t)B_SZ * 4);
  (void)ws_size; (void)n_in; (void)in_sizes; (void)out_size;

  // 1) pack inputs to bf16 (weights transposed to B^T layout)
  cvt_f32_bf16<<<2048, 256, 0, stream>>>(x, xb, (size_t)B_SZ * D_SZ);
  cvt_f32_bf16<<<512, 256, 0, stream>>>(E, eb, (size_t)H_SZ * K_SZ * D_SZ);
  transpose_cvt<<<4096, 256, 0, stream>>>(W1, w1t, D_SZ, HID_SZ, (size_t)H_SZ * D_SZ * HID_SZ);
  transpose_cvt<<<4096, 256, 0, stream>>>(W2, w2t, HID_SZ, D_SZ, (size_t)H_SZ * HID_SZ * D_SZ);
  avz_kernel<<<(K_SZ * D_SZ) / 256, 256, 0, stream>>>(E, avzf, avzb, K_SZ * D_SZ);

  // 2) row norms
  rownorm_bf16<<<B_SZ / 8, 256, 0, stream>>>(xb, xnrm, D_SZ);
  rownorm_bf16<<<(H_SZ * K_SZ) / 8, 256, 0, stream>>>(eb, enrm, D_SZ);
  rownorm_bf16<<<K_SZ / 8, 256, 0, stream>>>(avzb, aznrm, D_SZ);

  // 3) h = relu(x @ W1 + b1)  -> bf16 [H,B,HID]
  gemm_bf16_wmma<0><<<dim3(HID_SZ / 64, B_SZ / 64, H_SZ), 128, 0, stream>>>(
      xb, w1t, nullptr, b1, nullptr, hb,
      D_SZ, D_SZ, D_SZ, HID_SZ,
      0, (size_t)HID_SZ * D_SZ, (size_t)B_SZ * HID_SZ, 0, HID_SZ);

  // 4) z = h @ W2 + b2  -> bf16 [H,B,D]
  gemm_bf16_wmma<1><<<dim3(D_SZ / 64, B_SZ / 64, H_SZ), 128, 0, stream>>>(
      hb, w2t, nullptr, b2, nullptr, zb,
      HID_SZ, HID_SZ, HID_SZ, D_SZ,
      (size_t)B_SZ * HID_SZ, (size_t)D_SZ * HID_SZ, (size_t)B_SZ * D_SZ, 0, D_SZ);

  // 5) ||z||^2 per (h,b)
  rownorm_bf16<<<(H_SZ * B_SZ) / 8, 256, 0, stream>>>(zb, znrm, D_SZ);

  // 6) distances[h,b,k] = ||z||^2 + ||e||^2 - 2 z.e
  gemm_bf16_wmma<2><<<dim3(K_SZ / 64, B_SZ / 64, H_SZ), 128, 0, stream>>>(
      zb, eb, znrm, enrm, dist, nullptr,
      D_SZ, D_SZ, D_SZ, K_SZ,
      (size_t)B_SZ * D_SZ, (size_t)K_SZ * D_SZ, (size_t)B_SZ * K_SZ, B_SZ, K_SZ);

  // 7) idx = argmin_k mean_h distances ; vq_z = mean_h E[h, idx]
  argmin_kernel<<<B_SZ / 8, 256, 0, stream>>>(dist, idxp);
  vq_gather<<<(B_SZ * D_SZ) / 256, 256, 0, stream>>>(E, idxp, out0);

  // 8) all_vq_loss[k,b] = 1.25 * (||avz_k||^2 + ||x_b||^2 - 2 avz_k.x_b)
  gemm_bf16_wmma<3><<<dim3(B_SZ / 64, K_SZ / 64, 1), 128, 0, stream>>>(
      avzb, xb, aznrm, xnrm, loss, nullptr,
      D_SZ, D_SZ, D_SZ, B_SZ,
      0, 0, 0, 0, 0);

  // 9) all_vq_z[k,b,:] = avz[k,:]   (512 MB, non-temporal)
  {
    unsigned rowlen4 = D_SZ / 4;                        // 128
    unsigned perK = (unsigned)B_SZ * rowlen4;           // 524288
    unsigned total4 = (unsigned)K_SZ * perK;            // 33,554,432
    broadcast_avz<<<total4 / 256, 256, 0, stream>>>((const v4f*)avzf, (v4f*)avqz, rowlen4, perK);
  }
}
